// MultiCellLSTM_77068893159877
// MI455X (gfx1250) — matrix operand order repaired and checked by reading, not compiled, not run on hardware
//
#include <hip/hip_runtime.h>
#include <hip/hip_bf16.h>

// Problem dims
#define B_DIM 16384
#define E_DIM 128
#define H_DIM 256
#define A_DIM 20
#define K_DIM 384   // H + E

typedef __bf16 bf16_t;
typedef __attribute__((ext_vector_type(16))) __bf16 bf16x16;
typedef __attribute__((ext_vector_type(8)))  __bf16 bf16x8;
typedef __attribute__((ext_vector_type(8)))  float  v8f;

union AFrag { bf16x16 f; bf16x8 h[2]; };

__device__ __forceinline__ float sigmoid_(float x) {
    return 1.0f / (1.0f + __expf(-x));
}
__device__ __forceinline__ float tanh_(float x) {
    // numerically safe tanh via exp(-2|x|)
    float ax = fabsf(x);
    float e  = __expf(-2.0f * ax);
    float t  = (1.0f - e) / (1.0f + e);
    return copysignf(t, x);
}

// ---------------------------------------------------------------------------
// Kernel 0: pack 8 weight matrices (fp32 row-major [K=384][N=256]) into bf16
// B-fragment layout for v_wmma_f32_16x16x32_bf16:
//   wp[((g*16 + tn)*12 + ks)*512 + lane*16 + e] = W_g[ks*32 + (lane/16)*16 + e][tn*16 + lane%16]
// (lane L holds N = L%16, K-offset = (L>=16)*16 + e within a 32-wide K step)
// ---------------------------------------------------------------------------
__global__ void mclstm_pack_w(const float* __restrict__ w0, const float* __restrict__ w1,
                              const float* __restrict__ w2, const float* __restrict__ w3,
                              const float* __restrict__ w4, const float* __restrict__ w5,
                              const float* __restrict__ w6, const float* __restrict__ w7,
                              bf16_t* __restrict__ wp) {
    int idx = blockIdx.x * 256 + threadIdx.x;            // 0 .. 786431
    int e    = idx & 15;
    int lane = (idx >> 4) & 31;
    int rest = idx >> 9;
    int ks   = rest % 12;
    int t    = rest / 12;                                // g*16 + tn
    int tn   = t & 15;
    int g    = t >> 4;
    int k = ks * 32 + (lane >> 4) * 16 + e;
    int n = tn * 16 + (lane & 15);
    const float* ws[8] = {w0, w1, w2, w3, w4, w5, w6, w7};
    wp[idx] = (bf16_t)(ws[g][k * H_DIM + n]);
}

// ---------------------------------------------------------------------------
// Kernel 1: per-row attention + build bf16 operands hx1=[h,x1], hx2=[h,x2m].
// One wave32 per row; butterfly reductions for dot products.
// ---------------------------------------------------------------------------
__global__ __launch_bounds__(256)
void mclstm_attn(const float* __restrict__ x1, const float* __restrict__ x2,
                 const float* __restrict__ h,  const float* __restrict__ Wa,
                 bf16_t* __restrict__ hx1, bf16_t* __restrict__ hx2) {
    int row  = (blockIdx.x * 256 + threadIdx.x) >> 5;    // one row per wave
    int lane = threadIdx.x & 31;

    // base = [h, x1] . Wa[0:384]
    float t = 0.0f;
    for (int k = lane; k < K_DIM; k += 32) {
        float v = (k < H_DIM) ? h[(size_t)row * H_DIM + k]
                              : x1[(size_t)row * E_DIM + (k - H_DIM)];
        t += v * Wa[k];
    }
    #pragma unroll
    for (int off = 16; off; off >>= 1) t += __shfl_xor(t, off, 32);
    float base = t;

    // scores[a] = base + x2[row,a,:] . Wa[384:512]
    float sc[A_DIM];
    const float* x2row = x2 + (size_t)row * A_DIM * E_DIM;
    #pragma unroll 4
    for (int a = 0; a < A_DIM; a++) {
        float s = 0.0f;
        const float* xp = x2row + a * E_DIM;
        for (int k = lane; k < E_DIM; k += 32) s += xp[k] * Wa[K_DIM + k];
        #pragma unroll
        for (int off = 16; off; off >>= 1) s += __shfl_xor(s, off, 32);
        sc[a] = s + base;
    }

    // softmax over the 20 scores (redundant per lane, all lanes identical)
    float mx = sc[0];
    #pragma unroll
    for (int a = 1; a < A_DIM; a++) mx = fmaxf(mx, sc[a]);
    float se = 0.0f;
    #pragma unroll
    for (int a = 0; a < A_DIM; a++) { sc[a] = __expf(sc[a] - mx); se += sc[a]; }
    float inv = 1.0f / se;

    // x2m and E-part of operands
    for (int k = lane; k < E_DIM; k += 32) {
        float acc = 0.0f;
        #pragma unroll
        for (int a = 0; a < A_DIM; a++) acc += sc[a] * x2row[a * E_DIM + k];
        acc *= inv;
        hx2[(size_t)row * K_DIM + H_DIM + k] = (bf16_t)acc;
        hx1[(size_t)row * K_DIM + H_DIM + k] = (bf16_t)x1[(size_t)row * E_DIM + k];
    }
    // H-part (shared by both operands)
    for (int k = lane; k < H_DIM; k += 32) {
        bf16_t hv = (bf16_t)h[(size_t)row * H_DIM + k];
        hx1[(size_t)row * K_DIM + k] = hv;
        hx2[(size_t)row * K_DIM + k] = hv;
    }
}

// ---------------------------------------------------------------------------
// Kernel 2: fused 8-way GEMM ([B,384]x[384,256] per gate, bf16 WMMA, f32 acc)
// + LSTM epilogue.
//   - Block = 8 waves sharing one N-tile (tn); each wave owns a 32x16 output
//     tile (2 M-tiles), 16 f32 accumulators.
//   - Weights for this tn are staged into LDS in two 48KB phases (gates 0-3,
//     then 4-7); WMMA B fragments are fed from ds_load instead of L2.
// Gate order g: 0=W1_f 1=W2_f 2=W1_i 3=W2_i 4=W1_c 5=W2_c 6=W1_o 7=W2_o
// (even g -> A = hx1, odd g -> A = hx2).  acc index = g*2 + mt.
// ---------------------------------------------------------------------------
__global__ __launch_bounds__(256)
void mclstm_gemm(const bf16_t* __restrict__ hx1, const bf16_t* __restrict__ hx2,
                 const bf16_t* __restrict__ wp,
                 const float* __restrict__ b1f, const float* __restrict__ b2f,
                 const float* __restrict__ b1i, const float* __restrict__ b2i,
                 const float* __restrict__ b1c, const float* __restrict__ b2c,
                 const float* __restrict__ b1o, const float* __restrict__ b2o,
                 const float* __restrict__ c1in, const float* __restrict__ c2in,
                 float* __restrict__ out) {
    extern __shared__ __align__(32) bf16_t wlds[];       // 4 gates * 12 ks * 512 = 24576 elems (48KB)

    int lane = threadIdx.x & 31;
    int wave = threadIdx.x >> 5;                         // 0..7
    int tn   = blockIdx.x & 15;                          // N tile, shared by block
    int tm2  = (blockIdx.x >> 4) * 8 + wave;             // 32-row M tile (0..511)

    int mbase = tm2 * 32;
    int mrow0 = mbase + (lane & 15);                     // A row, M-tile 0
    int mrow1 = mrow0 + 16;                              // A row, M-tile 1
    int khalf = lane >> 4;                               // A K-half select

    v8f acc[16];
    #pragma unroll
    for (int q = 0; q < 16; q++) acc[q] = (v8f){0,0,0,0,0,0,0,0};

    const bf16_t* a1r0 = hx1 + (size_t)mrow0 * K_DIM;
    const bf16_t* a1r1 = hx1 + (size_t)mrow1 * K_DIM;
    const bf16_t* a2r0 = hx2 + (size_t)mrow0 * K_DIM;
    const bf16_t* a2r1 = hx2 + (size_t)mrow1 * K_DIM;

    // Per-gate weight slice for this tn: 12*512 elems, global stride between
    // gates = 16*12*512 = 98304 elems.
    const bf16_t* wtn = wp + (size_t)tn * 12 * 512;

    #pragma unroll
    for (int half = 0; half < 2; half++) {
        const int gbase = half * 4;
        __syncthreads();                                 // WAR guard vs previous phase
        // stage 4 gates (3072 chunks of 8 bf16 = 48KB) cooperatively
        for (int c = threadIdx.x; c < 3072; c += 256) {
            int o = c * 8;                               // element offset in LDS
            int g = o / 6144;                            // 0..3 (local gate)
            int r = o - g * 6144;
            *(bf16x8*)(wlds + o) =
                *(const bf16x8*)(wtn + (size_t)(gbase + g) * 98304 + r);
        }
        __syncthreads();

        for (int ks = 0; ks < 12; ks++) {
            // 16-bit A fragment 16x32: lane holds one contiguous 16-elem run (K = khalf*16 + e)
            AFrag fa1m0, fa1m1, fa2m0, fa2m1;
            int o0 = ks * 32 + khalf * 16;
            int o1 = o0 + 8;
            fa1m0.h[0] = *(const bf16x8*)(a1r0 + o0);
            fa1m0.h[1] = *(const bf16x8*)(a1r0 + o1);
            fa1m1.h[0] = *(const bf16x8*)(a1r1 + o0);
            fa1m1.h[1] = *(const bf16x8*)(a1r1 + o1);
            fa2m0.h[0] = *(const bf16x8*)(a2r0 + o0);
            fa2m0.h[1] = *(const bf16x8*)(a2r0 + o1);
            fa2m1.h[0] = *(const bf16x8*)(a2r1 + o0);
            fa2m1.h[1] = *(const bf16x8*)(a2r1 + o1);
            #pragma unroll
            for (int gg = 0; gg < 4; gg++) {
                const int g = gbase + gg;
                bf16x16 fb = *(const bf16x16*)(wlds + (gg * 12 + ks) * 512 + lane * 16);
                const bf16x16& fam0 = (g & 1) ? fa2m0.f : fa1m0.f;
                const bf16x16& fam1 = (g & 1) ? fa2m1.f : fa1m1.f;
                acc[g * 2 + 0] = __builtin_amdgcn_wmma_f32_16x16x32_bf16(
                    false, fam0, false, fb, (short)0, acc[g * 2 + 0], false, false);
                acc[g * 2 + 1] = __builtin_amdgcn_wmma_f32_16x16x32_bf16(
                    false, fam1, false, fb, (short)0, acc[g * 2 + 1], false, false);
            }
        }
    }

    // Epilogue. C/D layout: lane L, vgpr r -> M = r + 8*(L>=16), N = L%16.
    int col = tn * 16 + (lane & 15);
    float vb1f = b1f[col], vb2f = b2f[col];
    float vb1i = b1i[col], vb2i = b2i[col];
    float vb1c = b1c[col], vb2c = b2c[col];
    float vb1o = b1o[col], vb2o = b2o[col];
    const size_t BH = (size_t)B_DIM * H_DIM;

    #pragma unroll
    for (int mt = 0; mt < 2; mt++) {
        #pragma unroll
        for (int r = 0; r < 8; r++) {
            int m = mbase + mt * 16 + r + khalf * 8;
            size_t idx = (size_t)m * H_DIM + col;
            float f1  = sigmoid_(acc[0 * 2 + mt][r] + vb1f);
            float f2  = sigmoid_(acc[1 * 2 + mt][r] + vb2f);
            float i1  = sigmoid_(acc[2 * 2 + mt][r] + vb1i);
            float i2  = sigmoid_(acc[3 * 2 + mt][r] + vb2i);
            float c1t = tanh_(acc[4 * 2 + mt][r] + vb1c);
            float c2t = tanh_(acc[5 * 2 + mt][r] + vb2c);
            float c1n = f1 * c1in[idx] + i1 * c1t;
            float c2n = f2 * c2in[idx] + i2 * c2t;
            // softmax over {o1,o2} -> sigmoid(o1 - o2)
            float g1  = sigmoid_((acc[6 * 2 + mt][r] + vb1o) - (acc[7 * 2 + mt][r] + vb2o));
            float hn  = g1 * tanh_(c1n) + (1.0f - g1) * tanh_(c2n);
            out[idx]          = hn;
            out[BH + idx]     = c1n;
            out[2 * BH + idx] = c2n;
        }
    }
}

// ---------------------------------------------------------------------------
// Host launch
// ---------------------------------------------------------------------------
extern "C" void kernel_launch(void* const* d_in, const int* in_sizes, int n_in,
                              void* d_out, int out_size, void* d_ws, size_t ws_size,
                              hipStream_t stream) {
    // Input order: x1, x2, h, c1, c2, then (W1,W2,b1,b2) for f,i,c,o, then W_attn
    const float* x1 = (const float*)d_in[0];
    const float* x2 = (const float*)d_in[1];
    const float* h  = (const float*)d_in[2];
    const float* c1 = (const float*)d_in[3];
    const float* c2 = (const float*)d_in[4];
    const float* W1f = (const float*)d_in[5];  const float* W2f = (const float*)d_in[6];
    const float* b1f = (const float*)d_in[7];  const float* b2f = (const float*)d_in[8];
    const float* W1i = (const float*)d_in[9];  const float* W2i = (const float*)d_in[10];
    const float* b1i = (const float*)d_in[11]; const float* b2i = (const float*)d_in[12];
    const float* W1c = (const float*)d_in[13]; const float* W2c = (const float*)d_in[14];
    const float* b1c = (const float*)d_in[15]; const float* b2c = (const float*)d_in[16];
    const float* W1o = (const float*)d_in[17]; const float* W2o = (const float*)d_in[18];
    const float* b1o = (const float*)d_in[19]; const float* b2o = (const float*)d_in[20];
    const float* Wa  = (const float*)d_in[21];
    float* out = (float*)d_out;

    // Workspace layout (bf16): Wpack (1.5 MB), hx1 (12 MB), hx2 (12 MB)
    bf16_t* wp   = (bf16_t*)d_ws;                                   // 8*16*12*512 = 786432 elems
    bf16_t* hx1b = (bf16_t*)((char*)d_ws + (size_t)786432 * 2);     // B*384
    bf16_t* hx2b = hx1b + (size_t)B_DIM * K_DIM;

    // 0) pack weights to WMMA B-fragment layout (786432 elems)
    mclstm_pack_w<<<786432 / 256, 256, 0, stream>>>(W1f, W2f, W1i, W2i,
                                                    W1c, W2c, W1o, W2o, wp);
    // 1) attention + bf16 operand build (one wave per row, 8 waves/block)
    mclstm_attn<<<B_DIM / 8, 256, 0, stream>>>(x1, x2, h, Wa, hx1b, hx2b);
    // 2) fused 8-way bf16 WMMA GEMM (LDS-staged weights) + LSTM epilogue
    //    1024 blocks: tn = blk & 15, 8 waves x 2 M-tiles each -> 512 x 16 tiles
    mclstm_gemm<<<1024, 256, 48 * 1024, stream>>>(
        hx1b, hx2b, wp,
        b1f, b2f, b1i, b2i, b1c, b2c, b1o, b2o,
        c1, c2, out);
}
